// StandardEncoderLayer_90847148245358
// MI455X (gfx1250) — matrix operand order repaired and checked by reading, not compiled, not run on hardware
//
#include <hip/hip_runtime.h>
#include <hip/hip_bf16.h>
#include <math.h>

typedef __bf16 bf16_t;
typedef __attribute__((ext_vector_type(16))) __bf16 v16bf;
typedef __attribute__((ext_vector_type(8)))  float  v8f;
typedef int v4i __attribute__((vector_size(16)));

#define BB   2
#define SEQ  2048
#define DM   1024
#define NH   16
#define HD   64
#define DFFN 4096

// ---- CDNA5 async global->LDS path (gated; fallback = synchronous staging) --
#if defined(__has_builtin)
#if __has_builtin(__builtin_amdgcn_global_load_async_to_lds_b128)
#define HAVE_ASYNC_LDS 1
#endif
#endif
#ifndef HAVE_ASYNC_LDS
#define HAVE_ASYNC_LDS 0
#endif

#if HAVE_ASYNC_LDS
#if __has_builtin(__builtin_amdgcn_s_wait_asynccnt)
#define WAIT_ASYNC(n) __builtin_amdgcn_s_wait_asynccnt(n)
#else
#define WAIT_ASYNC(n) asm volatile("s_wait_asynccnt " #n ::: "memory")
#endif
typedef __attribute__((address_space(1))) v4i* gv4i_t;
typedef __attribute__((address_space(3))) v4i* lv4i_t;
#else
#define WAIT_ASYNC(n)
#endif

union BF16Frag { v16bf v; __bf16 e[16]; unsigned u[8]; };

// A-fragment K index for 16-bit WMMA A layout (16x32):
// VGPR j holds K pair; lanes 16-31 get +8 K offset.
__device__ __forceinline__ int kmapA(int j, int g) {
    return ((j & 3) * 2) + ((j >> 2) * 16) + g * 8;
}

// Stage 16 bytes global -> LDS (async DMA when available).
__device__ __forceinline__ void stage16(const bf16_t* g, bf16_t* l) {
#if HAVE_ASYNC_LDS
    __builtin_amdgcn_global_load_async_to_lds_b128(
        (gv4i_t)(v4i*)g, (lv4i_t)(v4i*)l, 0, 0);
#else
    *(uint4*)l = *(const uint4*)g;
#endif
}

// ---------------------------------------------------------------------------
// fp32 -> bf16 conversion
// ---------------------------------------------------------------------------
__global__ void cvt_f32_bf16_kernel(const float* __restrict__ src,
                                    bf16_t* __restrict__ dst, size_t n) {
    size_t i = (size_t)blockIdx.x * blockDim.x + threadIdx.x;
    if (i < n) dst[i] = (bf16_t)src[i];
}

// ---------------------------------------------------------------------------
// K re-layout: [B,S,D] bf16 -> [B,H,DH,S] bf16 (rows = contraction dim d)
// ---------------------------------------------------------------------------
__global__ void transpose_k_kernel(const bf16_t* __restrict__ Kn,
                                   bf16_t* __restrict__ Kt) {
    size_t idx = (size_t)blockIdx.x * blockDim.x + threadIdx.x;
    size_t total = (size_t)BB * SEQ * DM;
    if (idx >= total) return;
    int dmodel = (int)(idx % DM);
    size_t rest = idx / DM;
    int s = (int)(rest % SEQ);
    int b = (int)(rest / SEQ);
    int h = dmodel / HD;
    int d = dmodel % HD;
    Kt[(((size_t)(b * NH + h) * HD) + d) * SEQ + s] = Kn[idx];
}

// ---------------------------------------------------------------------------
// bf16 WMMA GEMM with double-buffered async-LDS staging of the shared B tile.
// C[M,N] = A[M,K]*B[K,N] + bias.  Block = 4 waves, tile 64x64, K-step 32.
// The A fragment is loaded at the top of the iteration so its global latency
// hides behind the barrier; all 4 B fragments are pulled from LDS into
// registers before the WMMA chain so the MMAs issue back-to-back.
// ---------------------------------------------------------------------------
#define BP 72

__global__ __launch_bounds__(128)
void gemm_bf16_kernel(const bf16_t* __restrict__ A, const bf16_t* __restrict__ Bm,
                      const float* __restrict__ bias,
                      float* __restrict__ outF, bf16_t* __restrict__ outB,
                      int M, int N, int K, int relu) {
    __shared__ bf16_t bs[2][32 * BP];

    int tid  = threadIdx.x;
    int lane = tid & 31;
    int wid  = tid >> 5;
    int g  = lane >> 4;
    int hn = lane & 15;
    int m0 = blockIdx.y * 64 + wid * 16;
    int n0 = blockIdx.x * 64;

    // cooperative B-tile copy: thread -> (row 0..31, 16-col segment)
    int crow = tid >> 2;
    int cseg = (tid & 3) * 16;

    auto stage_tile = [&](int buf, int kk) {
        const bf16_t* src = Bm + (size_t)(kk + crow) * N + n0 + cseg;
        bf16_t* dst = &bs[buf][crow * BP + cseg];
        stage16(src, dst);
        stage16(src + 8, dst + 8);
    };

    v8f z = {};
    v8f acc[4];
    #pragma unroll
    for (int t = 0; t < 4; ++t) acc[t] = z;

    stage_tile(0, 0);
    const bf16_t* arow0 = A + (size_t)(m0 + hn) * K;

    for (int kk = 0; kk < K; kk += 32) {
        int cur = (kk >> 5) & 1;
        bool more = (kk + 32 < K);

        // A fragment first: its LOADcnt latency overlaps barrier + async work
        BF16Frag a;
        #pragma unroll
        for (int j = 0; j < 8; ++j)
            a.u[j] = *(const unsigned*)(arow0 + kk + kmapA(j, g));
        if (more) __builtin_prefetch(arow0 + kk + 32, 0, 3);

        if (more) stage_tile(cur ^ 1, kk + 32);
        // own wave's copies for buffer `cur` complete (async done in order)
        if (more) { WAIT_ASYNC(2); } else { WAIT_ASYNC(0); }
        __syncthreads();   // all waves' copies of `cur` visible

        const bf16_t* brow = &bs[cur][hn];
        v16bf bfr[4];
        #pragma unroll
        for (int t = 0; t < 4; ++t)
            #pragma unroll
            for (int e = 0; e < 16; ++e)
                bfr[t][e] = brow[(e < 8 ? 8 * g + e : 8 + 8 * g + e) * BP + 16 * t];
        #pragma unroll
        for (int t = 0; t < 4; ++t)
            acc[t] = __builtin_amdgcn_wmma_f32_16x16x32_bf16(
                false, a.v, false, bfr[t], (short)0, acc[t], false, false);
        __syncthreads();   // reads of `cur` done before it is overwritten
    }

    #pragma unroll
    for (int t = 0; t < 4; ++t) {
        float bval = bias ? bias[n0 + 16 * t + hn] : 0.f;
        #pragma unroll
        for (int r = 0; r < 8; ++r) {
            int row = m0 + r + 8 * g;
            float v = acc[t][r] + bval;
            if (relu) v = fmaxf(v, 0.f);
            size_t idx = (size_t)row * N + n0 + 16 * t + hn;
            if (outF) outF[idx] = v;
            if (outB) outB[idx] = (bf16_t)v;
        }
    }
}

// ---------------------------------------------------------------------------
// Flash attention: wave = 16 query rows; online softmax; WMMA for QK^T + PV.
// Q [B,S,D] bf16, Kt [B,H,DH,S] bf16, V [B,S,D] bf16, mask [B,1,S,S] int.
// Output O [B,S,D] bf16.
// ---------------------------------------------------------------------------
__global__ __launch_bounds__(128)
void flash_attn_kernel(const bf16_t* __restrict__ Q, const bf16_t* __restrict__ Kt,
                       const bf16_t* __restrict__ V, const int* __restrict__ mask,
                       bf16_t* __restrict__ O) {
    __shared__ float ldsP[4 * 16 * 32];   // per-wave 16x32 P tile
    int lane = threadIdx.x & 31;
    int wid  = threadIdx.x >> 5;
    int g  = lane >> 4;
    int hn = lane & 15;
    int bh = blockIdx.x;
    int b  = bh / NH;
    int h  = bh % NH;
    int q0 = blockIdx.y * 64 + wid * 16;
    float* P = ldsP + wid * 512;

    // Q A-fragments for d = 0..31 and 32..63
    BF16Frag aq[2];
    const bf16_t* qrow = Q + (size_t)(b * SEQ + q0 + hn) * DM + h * HD;
    #pragma unroll
    for (int c = 0; c < 2; ++c)
        #pragma unroll
        for (int j = 0; j < 8; ++j)
            aq[c].u[j] = *(const unsigned*)(qrow + 32 * c + kmapA(j, g));

    float mrow[8], lrow[8], al[8];
    v8f z = {};
    v8f o[4];
    #pragma unroll
    for (int t = 0; t < 4; ++t) o[t] = z;
    #pragma unroll
    for (int r = 0; r < 8; ++r) { mrow[r] = -INFINITY; lrow[r] = 0.f; }

    const float scale = 0.125f;   // 1/sqrt(64)
    const bf16_t* ktBase = Kt + (size_t)bh * HD * SEQ;

    for (int kk = 0; kk < SEQ; kk += 32) {
        // ---- scores: 16q x 32k as two 16x16 fragments --------------------
        // batch all 4 K fragments first, then run the 4 WMMAs back-to-back
        v16bf kb[2][2];
        #pragma unroll
        for (int t = 0; t < 2; ++t)
            #pragma unroll
            for (int c = 0; c < 2; ++c)
                #pragma unroll
                for (int e = 0; e < 16; ++e)
                    kb[t][c][e] = ktBase[(size_t)(32 * c + (e < 8 ? 8 * g + e : 8 + 8 * g + e)) * SEQ + kk + 16 * t + hn];

        v8f sc[2];
        #pragma unroll
        for (int t = 0; t < 2; ++t) {
            v8f s = z;
            #pragma unroll
            for (int c = 0; c < 2; ++c)
                s = __builtin_amdgcn_wmma_f32_16x16x32_bf16(
                    false, aq[c].v, false, kb[t][c], (short)0, s, false, false);
            #pragma unroll
            for (int r = 0; r < 8; ++r) {
                int row = q0 + r + 8 * g;
                int col = kk + 16 * t + hn;
                int mv  = mask[(size_t)(b * SEQ + row) * SEQ + col];
                s[r] = mv ? s[r] * scale : -INFINITY;
            }
            sc[t] = s;
        }

        // ---- online softmax row statistics -------------------------------
        #pragma unroll
        for (int r = 0; r < 8; ++r) {
            float bm = fmaxf(sc[0][r], sc[1][r]);
            #pragma unroll
            for (int m = 1; m < 16; m <<= 1)
                bm = fmaxf(bm, __shfl_xor(bm, m, 32));
            float mn = fmaxf(mrow[r], bm);
            float alpha = (mn == -INFINITY) ? 1.f : __expf(mrow[r] - mn);
            float rs = 0.f;
            #pragma unroll
            for (int t = 0; t < 2; ++t) {
                float p = (mn == -INFINITY) ? 0.f : __expf(sc[t][r] - mn);
                sc[t][r] = p;
                rs += p;
            }
            #pragma unroll
            for (int m = 1; m < 16; m <<= 1)
                rs += __shfl_xor(rs, m, 32);
            lrow[r] = lrow[r] * alpha + rs;
            mrow[r] = mn;
            al[r] = alpha;
        }

        // ---- rescale running O accumulators ------------------------------
        #pragma unroll
        for (int t = 0; t < 4; ++t)
            #pragma unroll
            for (int r = 0; r < 8; ++r)
                o[t][r] *= al[r];

        // ---- P: C-layout -> LDS -> A-layout (bf16) ----------------------
        #pragma unroll
        for (int t = 0; t < 2; ++t)
            #pragma unroll
            for (int r = 0; r < 8; ++r)
                P[(r + 8 * g) * 32 + 16 * t + hn] = sc[t][r];
        __syncthreads();
        BF16Frag ap;
        #pragma unroll
        for (int j = 0; j < 8; ++j) {
            int k = kmapA(j, g);
            ap.e[2 * j]     = (bf16_t)P[hn * 32 + k];
            ap.e[2 * j + 1] = (bf16_t)P[hn * 32 + k + 1];
        }
        __syncthreads();

        // ---- O += P * V (batch V fragments, then chain WMMAs) ------------
        v16bf vb[4];
        #pragma unroll
        for (int t = 0; t < 4; ++t)
            #pragma unroll
            for (int e = 0; e < 16; ++e)
                vb[t][e] = V[(size_t)(b * SEQ + kk + (e < 8 ? 8 * g + e : 8 + 8 * g + e)) * DM + h * HD + 16 * t + hn];
        #pragma unroll
        for (int t = 0; t < 4; ++t)
            o[t] = __builtin_amdgcn_wmma_f32_16x16x32_bf16(
                false, ap.v, false, vb[t], (short)0, o[t], false, false);
    }

    // ---- normalize + store (fully-masked rows -> 0) ----------------------
    #pragma unroll
    for (int t = 0; t < 4; ++t)
        #pragma unroll
        for (int r = 0; r < 8; ++r) {
            float l = lrow[r];
            float v = (l > 0.f) ? o[t][r] / l : 0.f;
            O[(size_t)(b * SEQ + q0 + r + 8 * g) * DM + h * HD + 16 * t + hn] = (bf16_t)v;
        }
}

// ---------------------------------------------------------------------------
// Residual add + LayerNorm over D=1024 (one 256-thread block per row).
// ---------------------------------------------------------------------------
__global__ __launch_bounds__(256)
void add_ln_kernel(const float* __restrict__ a, const float* __restrict__ bres,
                   const float* __restrict__ gamma, const float* __restrict__ beta,
                   float* __restrict__ outF, bf16_t* __restrict__ outB) {
    __shared__ float ssum[256];
    __shared__ float ssq[256];
    int row = blockIdx.x;
    int tid = threadIdx.x;
    const float* ar = a    + (size_t)row * DM;
    const float* br = bres + (size_t)row * DM;

    float x[4];
    float lsum = 0.f;
    #pragma unroll
    for (int i = 0; i < 4; ++i) {
        int col = tid + 256 * i;
        x[i] = ar[col] + br[col];
        lsum += x[i];
    }
    ssum[tid] = lsum;
    __syncthreads();
    for (int s = 128; s > 0; s >>= 1) {
        if (tid < s) ssum[tid] += ssum[tid + s];
        __syncthreads();
    }
    float mu = ssum[0] * (1.f / DM);

    float lsq = 0.f;
    #pragma unroll
    for (int i = 0; i < 4; ++i) {
        float d = x[i] - mu;
        lsq += d * d;
    }
    ssq[tid] = lsq;
    __syncthreads();
    for (int s = 128; s > 0; s >>= 1) {
        if (tid < s) ssq[tid] += ssq[tid + s];
        __syncthreads();
    }
    float rstd = rsqrtf(ssq[0] * (1.f / DM) + 1e-5f);

    #pragma unroll
    for (int i = 0; i < 4; ++i) {
        int col = tid + 256 * i;
        float v = (x[i] - mu) * rstd * gamma[col] + beta[col];
        size_t idx = (size_t)row * DM + col;
        outF[idx] = v;
        if (outB) outB[idx] = (bf16_t)v;
    }
}

// ---------------------------------------------------------------------------
// Launch
// ---------------------------------------------------------------------------
extern "C" void kernel_launch(void* const* d_in, const int* in_sizes, int n_in,
                              void* d_out, int out_size, void* d_ws, size_t ws_size,
                              hipStream_t stream) {
    const float* x   = (const float*)d_in[0];
    const int*   msk = (const int*)  d_in[1];
    const float* Wq  = (const float*)d_in[2];
    const float* bq  = (const float*)d_in[3];
    const float* Wk  = (const float*)d_in[4];
    const float* bk  = (const float*)d_in[5];
    const float* Wv  = (const float*)d_in[6];
    const float* bv  = (const float*)d_in[7];
    const float* Wo  = (const float*)d_in[8];
    const float* bo  = (const float*)d_in[9];
    const float* W1  = (const float*)d_in[10];
    const float* bf1 = (const float*)d_in[11];
    const float* W2  = (const float*)d_in[12];
    const float* bf2 = (const float*)d_in[13];
    const float* g1  = (const float*)d_in[14];
    const float* b1  = (const float*)d_in[15];
    const float* g2  = (const float*)d_in[16];
    const float* b2  = (const float*)d_in[17];
    float* out = (float*)d_out;

    const size_t M = (size_t)BB * SEQ;              // 4096 rows
    char* ws = (char*)d_ws;
    size_t off = 0;
    auto alloc = [&](size_t bytes) -> void* {
        void* p = ws + off;
        off += (bytes + 255) & ~(size_t)255;
        return p;
    };

    bf16_t* xbf   = (bf16_t*)alloc(M * DM * 2);
    bf16_t* wqb   = (bf16_t*)alloc((size_t)DM * DM * 2);
    bf16_t* wkb   = (bf16_t*)alloc((size_t)DM * DM * 2);
    bf16_t* wvb   = (bf16_t*)alloc((size_t)DM * DM * 2);
    bf16_t* wob   = (bf16_t*)alloc((size_t)DM * DM * 2);
    bf16_t* w1b   = (bf16_t*)alloc((size_t)DM * DFFN * 2);
    bf16_t* w2b   = (bf16_t*)alloc((size_t)DFFN * DM * 2);
    bf16_t* qbf   = (bf16_t*)alloc(M * DM * 2);
    bf16_t* knbf  = (bf16_t*)alloc(M * DM * 2);
    bf16_t* ktbf  = (bf16_t*)alloc(M * DM * 2);
    bf16_t* vbf   = (bf16_t*)alloc(M * DM * 2);
    bf16_t* attnb = (bf16_t*)alloc(M * DM * 2);
    float*  projf = (float*) alloc(M * DM * 4);
    float*  inter = (float*) alloc(M * DM * 4);
    bf16_t* intb  = (bf16_t*)alloc(M * DM * 2);
    bf16_t* h1b   = (bf16_t*)alloc(M * DFFN * 2);
    float*  h2f   = (float*) alloc(M * DM * 4);
    (void)ws_size; (void)n_in; (void)in_sizes; (void)out_size;

    auto cvt = [&](const float* s, bf16_t* d, size_t n) {
        cvt_f32_bf16_kernel<<<dim3((unsigned)((n + 255) / 256)), dim3(256), 0, stream>>>(s, d, n);
    };
    cvt(x,  xbf, M * DM);
    cvt(Wq, wqb, (size_t)DM * DM);
    cvt(Wk, wkb, (size_t)DM * DM);
    cvt(Wv, wvb, (size_t)DM * DM);
    cvt(Wo, wob, (size_t)DM * DM);
    cvt(W1, w1b, (size_t)DM * DFFN);
    cvt(W2, w2b, (size_t)DFFN * DM);

    auto gemm = [&](const bf16_t* A, const bf16_t* B, const float* bias,
                    float* oF, bf16_t* oB, int m, int n, int k, int relu) {
        dim3 grid(n / 64, m / 64);
        gemm_bf16_kernel<<<grid, dim3(128), 0, stream>>>(A, B, bias, oF, oB, m, n, k, relu);
    };

    // QKV projections (bf16 outputs)
    gemm(xbf, wqb, bq, nullptr, qbf,  (int)M, DM, DM, 0);
    gemm(xbf, wkb, bk, nullptr, knbf, (int)M, DM, DM, 0);
    gemm(xbf, wvb, bv, nullptr, vbf,  (int)M, DM, DM, 0);

    // K -> [B,H,DH,S]
    {
        size_t n = M * DM;
        transpose_k_kernel<<<dim3((unsigned)((n + 255) / 256)), dim3(256), 0, stream>>>(knbf, ktbf);
    }

    // attention
    flash_attn_kernel<<<dim3(BB * NH, SEQ / 64), dim3(128), 0, stream>>>(qbf, ktbf, vbf, msk, attnb);

    // output projection (fp32 for residual)
    gemm(attnb, wob, bo, projf, nullptr, (int)M, DM, DM, 0);

    // residual + LN1 (fp32 + bf16 copy for FFN)
    add_ln_kernel<<<dim3((unsigned)M), dim3(256), 0, stream>>>(x, projf, g1, b1, inter, intb);

    // FFN
    gemm(intb, w1b, bf1, nullptr, h1b, (int)M, DFFN, DM, 1);   // ReLU, bf16
    gemm(h1b,  w2b, bf2, h2f, nullptr, (int)M, DM, DFFN, 0);

    // residual + LN2 -> output
    add_ln_kernel<<<dim3((unsigned)M), dim3(256), 0, stream>>>(inter, h2f, g2, b2, out, nullptr);
}
